// OverlapGATv2_75960791597985
// MI455X (gfx1250) — matrix operand order repaired and checked by reading, not compiled, not run on hardware
//
#include <hip/hip_runtime.h>

// ---------------------------------------------------------------------------
// Types for CDNA5 WMMA (wave32): v_wmma_f32_16x16x32_bf16
// ---------------------------------------------------------------------------
typedef __attribute__((ext_vector_type(16))) __bf16 v16bf;
typedef __attribute__((ext_vector_type(8)))  float  v8f;

__device__ __forceinline__ unsigned short f2bf(float f) {
  unsigned int u = __float_as_uint(f);
  unsigned int r = u + 0x7FFFu + ((u >> 16) & 1u);   // round-to-nearest-even
  return (unsigned short)(r >> 16);
}

__device__ __forceinline__ float waveSum(float v) {
#pragma unroll
  for (int o = 16; o > 0; o >>= 1) v += __shfl_xor(v, o, 32);
  return v;
}
__device__ __forceinline__ float waveMax(float v) {
#pragma unroll
  for (int o = 16; o > 0; o >>= 1) v = fmaxf(v, __shfl_xor(v, o, 32));
  return v;
}

// ---------------------------------------------------------------------------
// Generic strided bf16 WMMA GEMM:  C[M,N] (fp32) = act( A[M,K] * B[K,N] + bias )
// One wave computes a 32x32 output tile = 2x2 WMMA 16x16 tiles (4 WMMAs per
// K-step of 32). Main loop is in pointer-induction form. All operand layouts
// in this pipeline are K-contiguous, so the runtime path is the dword-packed
// <true,true> loop: 32 global_load_b32 + 4 WMMAs per K-step.
// ---------------------------------------------------------------------------
struct GemmParams {
  const unsigned short* A; long long sAM, sAK, sAB;   // A strides (M, K, batch)
  const unsigned short* B; long long sBK, sBN, sBB;   // B strides (K, N, batch)
  float*          C;   long long sCM, sCN, sCB;       // fp32 out (nullable)
  unsigned short* Cbf; long long sDM, sDN, sDB;       // bf16 out (nullable)
  const float* bias;                                  // per-N bias (nullable)
  int M, N, K;
  int relu;
  int atomicAccum;                                    // split-K accumulation
};

__device__ __forceinline__ v8f wmma_bf16(const v16bf& a, const v16bf& b, const v8f& c) {
  return __builtin_amdgcn_wmma_f32_16x16x32_bf16(false, a, false, b, (short)0, c,
                                                 false, false);
}

// A fragment (16-bit A, 16x32), contiguous k = 16*half + s order. 'ap' already includes the
// half*16 lane-group K offset. CONTIG=true -> dword-packed loads.
template <bool CONTIG>
__device__ __forceinline__ v16bf fragA(const unsigned short* ap, long long sK) {
  union { v16bf v; unsigned int u32[8]; unsigned short u16[16]; } r;
  if (CONTIG) {
#pragma unroll
    for (int j = 0; j < 8; ++j) {
      int rel = 2 * j;
      r.u32[j] = *(const unsigned int*)(ap + rel);
    }
  } else {
#pragma unroll
    for (int j = 0; j < 8; ++j) {
      int rel = 2 * j;
      r.u16[2 * j]     = ap[(long long)rel * sK];
      r.u16[2 * j + 1] = ap[(long long)(rel + 1) * sK];
    }
  }
  return r.v;
}

// B fragment (16-bit B, 32x16). 'bp' already includes the half*16 K offset.
template <bool CONTIG>
__device__ __forceinline__ v16bf fragB(const unsigned short* bp, long long sK) {
  union { v16bf v; unsigned int u32[8]; unsigned short u16[16]; } r;
  if (CONTIG) {
#pragma unroll
    for (int j = 0; j < 8; ++j) r.u32[j] = *(const unsigned int*)(bp + 2 * j);
  } else {
#pragma unroll
    for (int j = 0; j < 8; ++j) {
      r.u16[2 * j]     = bp[(long long)(2 * j) * sK];
      r.u16[2 * j + 1] = bp[(long long)(2 * j + 1) * sK];
    }
  }
  return r.v;
}

// Tail loaders (single partial K-step): unconditional loads with clamped
// indices + branch-free zero selects (v_cndmask, no EXEC branches).
__device__ __forceinline__ v16bf tailA(const unsigned short* rowp, long long sK,
                                       int k0, int half, int Kend) {
  union { v16bf v; unsigned short u16[16]; } r;
#pragma unroll
  for (int j = 0; j < 8; ++j) {
    int kk = k0 + 2 * j + half * 16;
    int k0c = (kk     < Kend) ? kk     : 0;
    int k1c = (kk + 1 < Kend) ? kk + 1 : 0;
    unsigned short v0 = rowp[(long long)k0c * sK];
    unsigned short v1 = rowp[(long long)k1c * sK];
    r.u16[2 * j]     = (kk     < Kend) ? v0 : (unsigned short)0;
    r.u16[2 * j + 1] = (kk + 1 < Kend) ? v1 : (unsigned short)0;
  }
  return r.v;
}
__device__ __forceinline__ v16bf tailB(const unsigned short* colp, long long sK,
                                       int k0, int half, int Kend) {
  union { v16bf v; unsigned short u16[16]; } r;
#pragma unroll
  for (int j = 0; j < 8; ++j) {
    int kk = k0 + half * 16 + 2 * j;
    int k0c = (kk     < Kend) ? kk     : 0;
    int k1c = (kk + 1 < Kend) ? kk + 1 : 0;
    unsigned short v0 = colp[(long long)k0c * sK];
    unsigned short v1 = colp[(long long)k1c * sK];
    r.u16[2 * j]     = (kk     < Kend) ? v0 : (unsigned short)0;
    r.u16[2 * j + 1] = (kk + 1 < Kend) ? v1 : (unsigned short)0;
  }
  return r.v;
}

// Main K loop, specialized on operand contiguity; dispatch once per kernel.
template <bool AC, bool BC>
__device__ __forceinline__ void gemmLoop(
    const unsigned short* rowp0, const unsigned short* rowp1,
    const unsigned short* colp0, const unsigned short* colp1,
    long long sA, long long sB, int kBegin, int kFull, int half,
    v8f& a00, v8f& a01, v8f& a10, v8f& a11) {
  const unsigned short* ap0 = rowp0 + (long long)(kBegin + half * 16) * sA;
  const unsigned short* ap1 = rowp1 + (long long)(kBegin + half * 16) * sA;
  const unsigned short* bp0 = colp0 + (long long)(kBegin + half * 16) * sB;
  const unsigned short* bp1 = colp1 + (long long)(kBegin + half * 16) * sB;
  const long long stepA = 32 * sA, stepB = 32 * sB;
  for (int k0 = kBegin; k0 + 32 <= kFull; k0 += 32) {
    __builtin_prefetch(ap0 + stepA, 0, 0);   // global_prefetch_b8 on gfx1250
    v16bf fa0 = fragA<AC>(ap0, sA);
    v16bf fa1 = fragA<AC>(ap1, sA);
    v16bf fb0 = fragB<BC>(bp0, sB);
    v16bf fb1 = fragB<BC>(bp1, sB);
    a00 = wmma_bf16(fa0, fb0, a00);
    a01 = wmma_bf16(fa0, fb1, a01);
    a10 = wmma_bf16(fa1, fb0, a10);
    a11 = wmma_bf16(fa1, fb1, a11);
    ap0 += stepA; ap1 += stepA; bp0 += stepB; bp1 += stepB;
  }
}

// C/D layout (ISA 7.12.2): VGPR i, lanes 0-15 -> M=i, lanes 16-31 -> M=8+i
__device__ __forceinline__ void storeTile(const v8f& acc, int mBase, int nBase,
                                          const GemmParams& p, float* Cb,
                                          unsigned short* Db, int lane) {
  int half = lane >> 4, lr = lane & 15;
  int n = nBase + lr;
  if (n >= p.N) return;
  float bias = (p.bias && !p.atomicAccum) ? p.bias[n] : 0.0f;
#pragma unroll
  for (int i = 0; i < 8; ++i) {
    int m = mBase + i + 8 * half;
    if (m >= p.M) continue;
    float v = acc[i] + bias;
    if (p.relu) v = fmaxf(v, 0.0f);
    if (Cb) {
      long long ci = (long long)m * p.sCM + (long long)n * p.sCN;
      if (p.atomicAccum) atomicAdd(&Cb[ci], v); else Cb[ci] = v;
    }
    if (Db) Db[(long long)m * p.sDM + (long long)n * p.sDN] = f2bf(v);
  }
}

__global__ void gemm_wmma_bf16(GemmParams p, int splitK) {
  int lane = threadIdx.x & 31;
  int half = lane >> 4, lr = lane & 15;
  int m0 = blockIdx.x * 32;
  int n0 = blockIdx.y * 32;
  int bz = blockIdx.z;
  int batch = bz / splitK, ks = bz % splitK;

  int kSteps = (p.K + 31) / 32;
  int kPer   = (kSteps + splitK - 1) / splitK;
  int kBegin = ks * kPer * 32;
  int kEnd   = (ks + 1) * kPer * 32; if (kEnd > p.K) kEnd = p.K;
  int kFull  = kBegin + ((kEnd > kBegin) ? ((kEnd - kBegin) / 32) * 32 : 0);

  const unsigned short* A = p.A + (long long)batch * p.sAB;
  const unsigned short* B = p.B + (long long)batch * p.sBB;

  // Clamped row/col base pointers (clamped rows read valid memory; results
  // are masked at store time).
  int mA0 = m0 + lr;      if (mA0 >= p.M) mA0 = p.M - 1;
  int mA1 = m0 + 16 + lr; if (mA1 >= p.M) mA1 = p.M - 1;
  int nB0 = n0 + lr;      if (nB0 >= p.N) nB0 = p.N - 1;
  int nB1 = n0 + 16 + lr; if (nB1 >= p.N) nB1 = p.N - 1;
  const unsigned short* rowp0 = A + (long long)mA0 * p.sAM;
  const unsigned short* rowp1 = A + (long long)mA1 * p.sAM;
  const unsigned short* colp0 = B + (long long)nB0 * p.sBN;
  const unsigned short* colp1 = B + (long long)nB1 * p.sBN;

  v8f a00 = {}, a01 = {}, a10 = {}, a11 = {};

  bool ac = (p.sAK == 1), bc = (p.sBK == 1);
  if (ac && bc)
    gemmLoop<true,  true >(rowp0, rowp1, colp0, colp1, 1, 1, kBegin, kFull, half,
                           a00, a01, a10, a11);
  else if (ac)
    gemmLoop<true,  false>(rowp0, rowp1, colp0, colp1, 1, p.sBK, kBegin, kFull, half,
                           a00, a01, a10, a11);
  else if (bc)
    gemmLoop<false, true >(rowp0, rowp1, colp0, colp1, p.sAK, 1, kBegin, kFull, half,
                           a00, a01, a10, a11);
  else
    gemmLoop<false, false>(rowp0, rowp1, colp0, colp1, p.sAK, p.sBK, kBegin, kFull,
                           half, a00, a01, a10, a11);

  if (kFull < kEnd) {        // single tail K-step, branch-free masked loads
    v16bf fa0 = tailA(rowp0, p.sAK, kFull, half, kEnd);
    v16bf fa1 = tailA(rowp1, p.sAK, kFull, half, kEnd);
    v16bf fb0 = tailB(colp0, p.sBK, kFull, half, kEnd);
    v16bf fb1 = tailB(colp1, p.sBK, kFull, half, kEnd);
    a00 = wmma_bf16(fa0, fb0, a00);
    a01 = wmma_bf16(fa0, fb1, a01);
    a10 = wmma_bf16(fa1, fb0, a10);
    a11 = wmma_bf16(fa1, fb1, a11);
  }

  float*          Cb = p.C   ? p.C   + (long long)batch * p.sCB : nullptr;
  unsigned short* Db = p.Cbf ? p.Cbf + (long long)batch * p.sDB : nullptr;
  storeTile(a00, m0,      n0,      p, Cb, Db, lane);
  storeTile(a01, m0,      n0 + 16, p, Cb, Db, lane);
  storeTile(a10, m0 + 16, n0,      p, Cb, Db, lane);
  storeTile(a11, m0 + 16, n0 + 16, p, Cb, Db, lane);
}

// ---------------------------------------------------------------------------
// Direct NCHW conv, kernel (kh,1), stride (sh,1), VALID, ReLU. fp32.
// Used for conv1..conv5 only. Optional bf16 output is written POSITION-MAJOR
// ([b][ho][wd][co], channel contiguous) so the next GEMM's B is K-contiguous.
// ---------------------------------------------------------------------------
__global__ void conv_relu(const float* __restrict__ x, const float* __restrict__ w,
                          float* y, unsigned short* ybfT,
                          int Bn, int Ci, int Hi, int Wd, int Co, int kh, int sh, int Ho) {
  long long idx = (long long)blockIdx.x * blockDim.x + threadIdx.x;
  long long total = (long long)Bn * Co * Ho * Wd;
  if (idx >= total) return;
  int wd = (int)(idx % Wd); long long t = idx / Wd;
  int ho = (int)(t % Ho); t /= Ho;
  int co = (int)(t % Co); int b = (int)(t / Co);
  const float* wb = w + (long long)co * Ci * kh;             // OIHW, W=1
  const float* xb = x + (long long)b * Ci * Hi * Wd;
  float acc = 0.0f;
  for (int ci = 0; ci < Ci; ++ci)
    for (int k = 0; k < kh; ++k)
      acc += wb[ci * kh + k] * xb[((long long)ci * Hi + ho * sh + k) * Wd + wd];
  acc = fmaxf(acc, 0.0f);
  if (y)    y[idx] = acc;
  if (ybfT) ybfT[(((long long)b * Ho + ho) * Wd + wd) * Co + co] = f2bf(acc);
}

__global__ void f32_to_bf16_k(const float* __restrict__ s, unsigned short* __restrict__ d,
                              long long n) {
  long long i = (long long)blockIdx.x * blockDim.x + threadIdx.x;
  if (i < n) d[i] = f2bf(s[i]);
}

// Transposed conversion: src is [K,N] row-major fp32, dst is [N,K] bf16.
__global__ void f32_to_bf16_T(const float* __restrict__ s, unsigned short* __restrict__ d,
                              int K, int N) {
  long long i = (long long)blockIdx.x * blockDim.x + threadIdx.x;
  if (i >= (long long)K * N) return;
  int n = (int)(i % N), k = (int)(i / N);
  d[(long long)n * K + k] = f2bf(s[i]);
}

__global__ void zero_f32(float* p, long long n) {
  long long i = (long long)blockIdx.x * blockDim.x + threadIdx.x;
  if (i < n) p[i] = 0.0f;
}

// ---------------------------------------------------------------------------
// GATv2 edge softmax + aggregation. Ring graph: for dst d the incoming edges
// are src = (d - off) mod 900 for off in {-15..-1, 1..14} plus self loop.
// One wave32 per (sample, node): lane j owns edge j (30 edges).
// H = sum_j alpha_j * XL[src_j] + XRES[d] + bias   (written as bf16)
// ---------------------------------------------------------------------------
__global__ void gatv2_edge(const float* __restrict__ XL, const float* __restrict__ XR,
                           const float* __restrict__ XRES, const float* __restrict__ att,
                           const float* __restrict__ bias, unsigned short* __restrict__ Hbf,
                           int Fo) {
  const int Nn = 900;
  int d = blockIdx.x, b = blockIdx.y, lane = threadIdx.x;
  long long base = (long long)b * Nn * Fo;
  const float* xrd = XR + base + (long long)d * Fo;

  float e = -3.4e38f;
  int sIdx = d;
  if (lane < 30) {
    int off = (lane == 29) ? 0 : ((lane < 15) ? (lane - 15) : (lane - 14));
    int s = d - off; s = ((s % Nn) + Nn) % Nn;
    sIdx = s;
    const float* xls = XL + base + (long long)s * Fo;
    float acc = 0.0f;
    for (int f = 0; f < Fo; ++f) {
      float z = xls[f] + xrd[f];
      z = (z > 0.0f) ? z : 0.2f * z;                 // LeakyReLU(0.2)
      acc += z * att[f];
    }
    e = acc;
  }
  float m  = waveMax(e);
  float ex = (lane < 30) ? __expf(e - m) : 0.0f;
  float sd = waveSum(ex);
  float alpha = ex / sd;

  for (int f = lane; f < Fo; f += 32) {              // Fo multiple of 32
    float acc = 0.0f;
    for (int j = 0; j < 30; ++j) {
      float a = __shfl(alpha, j, 32);
      int  sj = __shfl(sIdx, j, 32);
      acc += a * XL[base + (long long)sj * Fo + f];
    }
    float v = acc + XRES[base + (long long)d * Fo + f] + bias[f];
    Hbf[base + (long long)d * Fo + f] = f2bf(v);
  }
}

// ---------------------------------------------------------------------------
// L2-normalize over 1024 channels of [B,1024,900]; emit bf16 in both layouts:
// fnk = [B,900,1024] (assignment GEMM A), ffn = [B,1024,900] (vlad GEMM A).
// ---------------------------------------------------------------------------
__global__ void norm_channels(const float* __restrict__ y, unsigned short* fnk,
                              unsigned short* ffn) {
  const int C = 1024, Wd = 900;
  int w = blockIdx.x, b = blockIdx.y, lane = threadIdx.x;
  long long base = (long long)b * C * Wd + w;
  float ss = 0.0f;
  for (int c = lane; c < C; c += 32) { float v = y[base + (long long)c * Wd]; ss += v * v; }
  ss = waveSum(ss);
  float inv = 1.0f / fmaxf(sqrtf(ss), 1e-12f);
  for (int c = lane; c < C; c += 32) {
    unsigned short h = f2bf(y[base + (long long)c * Wd] * inv);
    fnk[((long long)b * 900 + w) * 1024 + c] = h;
    ffn[((long long)b * 1024 + c) * 900 + w] = h;
  }
}

// Softmax over 64 clusters; fp32 act in [B,900,64] and bf16 act TRANSPOSED in
// [B,64,900] so the vlad GEMM's B operand is K(=node)-contiguous.
__global__ void softmax64(const float* __restrict__ logits, float* act,
                          unsigned short* actT) {
  int n = blockIdx.x, b = blockIdx.y, lane = threadIdx.x;
  long long base = ((long long)b * 900 + n) * 64;
  float v0 = logits[base + lane], v1 = logits[base + 32 + lane];
  float m = waveMax(fmaxf(v0, v1));
  float e0 = __expf(v0 - m), e1 = __expf(v1 - m);
  float s = waveSum(e0 + e1);
  float a0 = e0 / s, a1 = e1 / s;
  act[base + lane] = a0;       act[base + 32 + lane] = a1;
  actT[((long long)b * 64 + lane)      * 900 + n] = f2bf(a0);
  actT[((long long)b * 64 + lane + 32) * 900 + n] = f2bf(a1);
}

__global__ void colsum_act(const float* __restrict__ act, float* a_sum) {
  int k = blockIdx.x, b = blockIdx.y, lane = threadIdx.x;
  float s = 0.0f;
  for (int n = lane; n < 900; n += 32) s += act[((long long)b * 900 + n) * 64 + k];
  s = waveSum(s);
  if (lane == 0) a_sum[b * 64 + k] = s;
}

__global__ void vlad_sub_cnorm(float* vlad, const float* __restrict__ a_sum,
                               const float* __restrict__ cw2, float* cnorm) {
  int k = blockIdx.x, b = blockIdx.y, lane = threadIdx.x;
  float a = a_sum[b * 64 + k];
  float ss = 0.0f;
  for (int f = lane; f < 1024; f += 32) {
    long long i = ((long long)b * 1024 + f) * 64 + k;
    float v = vlad[i] - a * cw2[f * 64 + k];
    vlad[i] = v; ss += v * v;
  }
  ss = waveSum(ss);
  if (lane == 0) cnorm[b * 64 + k] = ss;
}

__global__ void vlad_intra(float* vlad, const float* __restrict__ cnorm, float* gnorm) {
  int k = blockIdx.x, b = blockIdx.y, lane = threadIdx.x;
  float inv = 1.0f / fmaxf(sqrtf(cnorm[b * 64 + k]), 1e-12f);
  float ss = 0.0f;
  for (int f = lane; f < 1024; f += 32) {
    long long i = ((long long)b * 1024 + f) * 64 + k;
    float v = vlad[i] * inv; vlad[i] = v; ss += v * v;
  }
  ss = waveSum(ss);
  if (lane == 0) atomicAdd(&gnorm[b], ss);
}

__global__ void vlad_global(const float* __restrict__ vlad, const float* __restrict__ gnorm,
                            unsigned short* vn) {
  long long i = (long long)blockIdx.x * blockDim.x + threadIdx.x;
  if (i >= (long long)32 * 65536) return;
  int b = (int)(i >> 16);
  float inv = 1.0f / fmaxf(sqrtf(gnorm[b]), 1e-12f);
  vn[i] = f2bf(vlad[i] * inv);
}

__global__ void bn_apply(const float* __restrict__ h, const float* __restrict__ g,
                         const float* __restrict__ be, float* v) {
  int i = blockIdx.x * blockDim.x + threadIdx.x;
  if (i >= 32 * 256) return;
  int o = i & 255;
  v[i] = h[i] * rsqrtf(1.0f + 1e-5f) * g[o] + be[o];
}

__global__ void gate_kernel(const float* __restrict__ v, const float* __restrict__ gw,
                            const float* __restrict__ gb, float* outpre) {
  int o = blockIdx.x, b = blockIdx.y, lane = threadIdx.x;
  float s = 0.0f;
  for (int j = lane; j < 256; j += 32) s += v[b * 256 + j] * gw[j * 256 + o];
  s = waveSum(s);
  float g = 1.0f / (1.0f + __expf(-(s + gb[o])));
  if (lane == 0) outpre[b * 256 + o] = v[b * 256 + o] * g;
}

__global__ void final_norm(const float* __restrict__ outpre, float* out) {
  int b = blockIdx.x, lane = threadIdx.x;
  float ss = 0.0f;
  for (int o = lane; o < 256; o += 32) { float x = outpre[b * 256 + o]; ss += x * x; }
  ss = waveSum(ss);
  float inv = 1.0f / fmaxf(sqrtf(ss), 1e-12f);
  for (int o = lane; o < 256; o += 32) out[b * 256 + o] = outpre[b * 256 + o] * inv;
}

// ---------------------------------------------------------------------------
// Host orchestration
// ---------------------------------------------------------------------------
static void launch_gemm(hipStream_t st, const GemmParams& p, int batch, int splitK) {
  dim3 g((p.M + 31) / 32, (p.N + 31) / 32, batch * splitK);
  gemm_wmma_bf16<<<g, 32, 0, st>>>(p, splitK);
}

extern "C" void kernel_launch(void* const* d_in, const int* in_sizes, int n_in,
                              void* d_out, int out_size, void* d_ws, size_t ws_size,
                              hipStream_t stream) {
  (void)in_sizes; (void)n_in; (void)out_size; (void)ws_size;
  const int Bn = 32;

  // ---- inputs (setup_inputs order) ----
  const float* x_l  = (const float*)d_in[0];
  const float* wc[12] = { nullptr,
    (const float*)d_in[1], (const float*)d_in[2], (const float*)d_in[3],
    (const float*)d_in[4], (const float*)d_in[5], (const float*)d_in[6],
    (const float*)d_in[7], (const float*)d_in[8], (const float*)d_in[9],
    (const float*)d_in[10], (const float*)d_in[11] };
  const float* w_last1 = (const float*)d_in[12];
  const float* w_last2 = (const float*)d_in[13];
  // g{1,2,3}: wl, bl, wr, br, att, res, bias
  const float* g_wl[3]   = { (const float*)d_in[14], (const float*)d_in[21], (const float*)d_in[28] };
  const float* g_bl[3]   = { (const float*)d_in[15], (const float*)d_in[22], (const float*)d_in[29] };
  const float* g_wr[3]   = { (const float*)d_in[16], (const float*)d_in[23], (const float*)d_in[30] };
  const float* g_br[3]   = { (const float*)d_in[17], (const float*)d_in[24], (const float*)d_in[31] };
  const float* g_att[3]  = { (const float*)d_in[18], (const float*)d_in[25], (const float*)d_in[32] };
  const float* g_res[3]  = { (const float*)d_in[19], (const float*)d_in[26], (const float*)d_in[33] };
  const float* g_bias[3] = { (const float*)d_in[20], (const float*)d_in[27], (const float*)d_in[34] };
  const float* nv_cw  = (const float*)d_in[35];
  const float* nv_cb  = (const float*)d_in[36];
  const float* nv_cw2 = (const float*)d_in[37];
  const float* nv_hw  = (const float*)d_in[38];
  const float* nv_bng = (const float*)d_in[39];
  const float* nv_bnb = (const float*)d_in[40];
  const float* nv_gw  = (const float*)d_in[41];
  const float* nv_gb  = (const float*)d_in[42];
  float* out = (float*)d_out;

  // ---- workspace bump allocator ----
  char* ws = (char*)d_ws;
  size_t off = 0;
  auto alloc = [&](size_t bytes) -> char* {
    char* p = ws + off;
    off = (off + bytes + 255) & ~(size_t)255;
    return p;
  };

  // conv chain fp32 arenas (reused later for GAT/feats buffers)
  float* X1 = (float*)alloc((size_t)32 * 16 * 60 * 900 * 4);   // 110.6 MB
  float* X2 = (float*)alloc((size_t)32 * 32 * 29 * 900 * 4);   // 106.9 MB
  float* X3 = (float*)alloc((size_t)32 * 64 * 14 * 900 * 4);   // 103.2 MB
  float* X4 = (float*)alloc((size_t)32 * 64 * 6 * 900 * 4);    //  44.2 MB
  // conv6..last1 intermediates, POSITION-MAJOR bf16 ([b][pos][ci])
  unsigned short* X5T  = (unsigned short*)alloc((size_t)32 * 3 * 900 * 128 * 2);
  unsigned short* X6T  = (unsigned short*)alloc((size_t)32 * 2 * 900 * 128 * 2);
  unsigned short* X7T  = (unsigned short*)alloc((size_t)32 * 900 * 128 * 2);
  unsigned short* X8T  = (unsigned short*)alloc((size_t)32 * 900 * 128 * 2);
  unsigned short* X9T  = (unsigned short*)alloc((size_t)32 * 900 * 128 * 2);
  unsigned short* X10T = (unsigned short*)alloc((size_t)32 * 900 * 128 * 2);
  unsigned short* X11T = (unsigned short*)alloc((size_t)32 * 900 * 128 * 2);
  unsigned short* NODES = (unsigned short*)alloc((size_t)32 * 900 * 256 * 2);
  unsigned short* H1BF  = (unsigned short*)alloc((size_t)32 * 900 * 256 * 2);
  unsigned short* H2BF  = (unsigned short*)alloc((size_t)32 * 900 * 512 * 2);
  unsigned short* H3BF  = (unsigned short*)alloc((size_t)32 * 900 * 512 * 2);
  float* XL2   = (float*)alloc((size_t)32 * 1024 * 900 * 4);   // 118 MB (w_last2 out)
  float* LOG   = (float*)alloc((size_t)32 * 900 * 64 * 4);
  float* ACT   = (float*)alloc((size_t)32 * 900 * 64 * 4);
  unsigned short* ACTT = (unsigned short*)alloc((size_t)32 * 64 * 900 * 2);  // [B,64,900]
  float* ASUM  = (float*)alloc((size_t)32 * 64 * 4);
  float* CNORM = (float*)alloc((size_t)32 * 64 * 4);
  float* GNORM = (float*)alloc((size_t)32 * 4);
  float* VLAD  = (float*)alloc((size_t)32 * 1024 * 64 * 4);
  unsigned short* VLADN = (unsigned short*)alloc((size_t)32 * 65536 * 2);
  float* HID   = (float*)alloc((size_t)32 * 256 * 4);
  float* VBN   = (float*)alloc((size_t)32 * 256 * 4);
  float* OPRE  = (float*)alloc((size_t)32 * 256 * 4);
  // bf16 weight copies. A-side weights stay [Co,Ci] row-major (K-contiguous);
  // B-side weights are stored TRANSPOSED [N,K] so fragments are K-contiguous.
  unsigned short* WC6  = (unsigned short*)alloc((size_t)128 * 128 * 2);
  unsigned short* WC7  = (unsigned short*)alloc((size_t)128 * 128 * 2);
  unsigned short* WC8  = (unsigned short*)alloc((size_t)128 * 128 * 2);
  unsigned short* WC9  = (unsigned short*)alloc((size_t)128 * 128 * 2);
  unsigned short* WC10 = (unsigned short*)alloc((size_t)128 * 128 * 2);
  unsigned short* WC11 = (unsigned short*)alloc((size_t)128 * 128 * 2);
  unsigned short* WL1B = (unsigned short*)alloc((size_t)256 * 128 * 2);
  unsigned short* WL2B = (unsigned short*)alloc((size_t)1024 * 512 * 2);
  unsigned short* GWT[3][3]; // [layer][wl,wr,res], transposed [Fo,Fi]
  const size_t gsz[3] = { (size_t)256 * 256, (size_t)256 * 512, (size_t)512 * 512 };
  for (int l = 0; l < 3; ++l)
    for (int j = 0; j < 3; ++j) GWT[l][j] = (unsigned short*)alloc(gsz[l] * 2);
  unsigned short* NVCWT = (unsigned short*)alloc((size_t)64 * 1024 * 2);    // [64,1024]
  unsigned short* NVHWT = (unsigned short*)alloc((size_t)256 * 65536 * 2);  // [256,65536]
  // GAT intermediates alias the dead conv arenas (X1..X3 are free by then)
  float* XLb   = X1;                    // [32,900,512] fp32 (58.9 MB <= 110.6)
  float* XRb   = X2;
  float* XRESb = X3;
  unsigned short* FNK = (unsigned short*)X1;  // feats [B,900,1024] bf16 (59 MB)
  unsigned short* FFN = (unsigned short*)X2;  // feats [B,1024,900] bf16

  auto conv = [&](const float* xin, const float* w, float* y, unsigned short* ybfT,
                  int Ci, int Hi, int Co, int kh, int sh, int Ho) {
    long long total = (long long)Bn * Co * Ho * 900;
    conv_relu<<<dim3((unsigned)((total + 255) / 256)), 256, 0, stream>>>(
        xin, w, y, ybfT, Bn, Ci, Hi, 900, Co, kh, sh, Ho);
  };
  auto cvt = [&](const float* s, unsigned short* d, long long n) {
    f32_to_bf16_k<<<dim3((unsigned)((n + 255) / 256)), 256, 0, stream>>>(s, d, n);
  };
  auto cvtT = [&](const float* s, unsigned short* d, int K, int N) {
    long long n = (long long)K * N;
    f32_to_bf16_T<<<dim3((unsigned)((n + 255) / 256)), 256, 0, stream>>>(s, d, K, N);
  };

  // ---- weight conversions ----
  cvt(wc[6], WC6, 128 * 128);   cvt(wc[7], WC7, 128 * 128);
  cvt(wc[8], WC8, 128 * 128);   cvt(wc[9], WC9, 128 * 128);
  cvt(wc[10], WC10, 128 * 128); cvt(wc[11], WC11, 128 * 128);
  cvt(w_last1, WL1B, 256 * 128);
  cvt(w_last2, WL2B, (long long)1024 * 512);
  const int gfi[3] = { 256, 256, 512 }, gfo[3] = { 256, 512, 512 };
  for (int l = 0; l < 3; ++l) {
    cvtT(g_wl[l],  GWT[l][0], gfi[l], gfo[l]);
    cvtT(g_wr[l],  GWT[l][1], gfi[l], gfo[l]);
    cvtT(g_res[l], GWT[l][2], gfi[l], gfo[l]);
  }
  cvtT(nv_cw, NVCWT, 1024, 64);
  cvtT(nv_hw, NVHWT, 65536, 256);

  // ---- conv1..conv5 direct fp32 (conv5 emits position-major bf16) ----
  conv(x_l, wc[1], X1, nullptr, 5, 64, 16, 5, 1, 60);
  conv(X1, wc[2], X2, nullptr, 16, 60, 32, 3, 2, 29);
  conv(X2, wc[3], X3, nullptr, 32, 29, 64, 3, 2, 14);
  conv(X3, wc[4], X4, nullptr, 64, 14, 64, 3, 2, 6);
  conv(X4, wc[5], nullptr, X5T, 64, 6, 128, 2, 2, 3);

  GemmParams p{};
  // ---- conv6 (stride-2 over H=3 -> h' in {0,2}), two GEMM slices ----
  // B = X5T [b][h][w][ci]: sBK=1 (ci contiguous), sBN=128 (next position).
  for (int hp = 0; hp < 2; ++hp) {
    p = GemmParams{};
    p.A = WC6;  p.sAM = 128; p.sAK = 1; p.sAB = 0;
    p.B = X5T + (size_t)(2 * hp) * 900 * 128; p.sBK = 1; p.sBN = 128;
    p.sBB = (long long)3 * 900 * 128;
    p.Cbf = X6T + (size_t)hp * 900 * 128; p.sDM = 1; p.sDN = 128;
    p.sDB = (long long)2 * 900 * 128;
    p.M = 128; p.N = 900; p.K = 128; p.relu = 1;
    launch_gemm(stream, p, Bn, 1);
  }
  // ---- conv7 (h=0 slice of X6T) and conv8..11, all 128x128 GEMMs ----
  const unsigned short* cin[5]   = { X6T, X7T, X8T, X9T, X10T };
  unsigned short*       cout5[5] = { X7T, X8T, X9T, X10T, X11T };
  const unsigned short* cw5[5]   = { WC7, WC8, WC9, WC10, WC11 };
  const long long inBat[5] = { (long long)2 * 900 * 128, 115200, 115200, 115200, 115200 };
  for (int s = 0; s < 5; ++s) {
    p = GemmParams{};
    p.A = cw5[s]; p.sAM = 128; p.sAK = 1; p.sAB = 0;
    p.B = cin[s]; p.sBK = 1; p.sBN = 128; p.sBB = inBat[s];
    p.Cbf = cout5[s]; p.sDM = 1; p.sDN = 128; p.sDB = 115200;
    p.M = 128; p.N = 900; p.K = 128; p.relu = 1;
    launch_gemm(stream, p, Bn, 1);
  }
  // ---- last1: 128->256, write as nodes [B,900,256] bf16 ----
  p = GemmParams{};
  p.A = WL1B; p.sAM = 128; p.sAK = 1; p.sAB = 0;
  p.B = X11T; p.sBK = 1; p.sBN = 128; p.sBB = 115200;
  p.Cbf = NODES; p.sDM = 1; p.sDN = 256; p.sDB = (long long)900 * 256;
  p.M = 256; p.N = 900; p.K = 128; p.relu = 1;
  launch_gemm(stream, p, Bn, 1);

  // ---- GATv2 layers (B = transposed weights [Fo,Fi], K-contiguous) ----
  const unsigned short* gin[3] = { NODES, H1BF, H2BF };
  unsigned short* gout[3] = { H1BF, H2BF, H3BF };
  for (int l = 0; l < 3; ++l) {
    const int Fi = gfi[l], Fo = gfo[l];
    float* dst[3] = { XLb, XRb, XRESb };
    const unsigned short* wmat[3] = { GWT[l][0], GWT[l][1], GWT[l][2] };
    const float* bvec[3] = { g_bl[l], g_br[l], nullptr };
    for (int t = 0; t < 3; ++t) {
      p = GemmParams{};
      p.A = gin[l]; p.sAM = Fi; p.sAK = 1; p.sAB = (long long)900 * Fi;
      p.B = wmat[t]; p.sBK = 1; p.sBN = Fi; p.sBB = 0;
      p.C = dst[t]; p.sCM = Fo; p.sCN = 1; p.sCB = (long long)900 * Fo;
      p.bias = bvec[t];
      p.M = 900; p.N = Fo; p.K = Fi;
      launch_gemm(stream, p, Bn, 1);
    }
    gatv2_edge<<<dim3(900, Bn), 32, 0, stream>>>(XLb, XRb, XRESb, g_att[l],
                                                 g_bias[l], gout[l], Fo);
  }

  // ---- last2: 512->1024 + ReLU -> [B,1024,900] fp32 ----
  p = GemmParams{};
  p.A = WL2B; p.sAM = 512; p.sAK = 1; p.sAB = 0;
  p.B = H3BF; p.sBK = 1; p.sBN = 512; p.sBB = (long long)900 * 512;
  p.C = XL2; p.sCM = 900; p.sCN = 1; p.sCB = (long long)1024 * 900;
  p.M = 1024; p.N = 900; p.K = 512; p.relu = 1;
  launch_gemm(stream, p, Bn, 1);

  // ---- channel L2 normalize -> feats bf16 (both layouts) ----
  norm_channels<<<dim3(900, Bn), 32, 0, stream>>>(XL2, FNK, FFN);

  // ---- NetVLAD assignment GEMM + softmax ----
  p = GemmParams{};
  p.A = FNK; p.sAM = 1024; p.sAK = 1; p.sAB = (long long)900 * 1024;
  p.B = NVCWT; p.sBK = 1; p.sBN = 1024; p.sBB = 0;
  p.C = LOG; p.sCM = 64; p.sCN = 1; p.sCB = (long long)900 * 64;
  p.bias = nv_cb;
  p.M = 900; p.N = 64; p.K = 1024;
  launch_gemm(stream, p, Bn, 1);
  softmax64<<<dim3(900, Bn), 32, 0, stream>>>(LOG, ACT, ACTT);
  colsum_act<<<dim3(64, Bn), 32, 0, stream>>>(ACT, ASUM);

  // ---- vlad GEMM: feats^T @ act, K=900 (tail-masked) ----
  p = GemmParams{};
  p.A = FFN; p.sAM = 900; p.sAK = 1; p.sAB = (long long)1024 * 900;
  p.B = ACTT; p.sBK = 1; p.sBN = 900; p.sBB = (long long)64 * 900;
  p.C = VLAD; p.sCM = 64; p.sCN = 1; p.sCB = (long long)1024 * 64;
  p.M = 1024; p.N = 64; p.K = 900;
  launch_gemm(stream, p, Bn, 1);

  vlad_sub_cnorm<<<dim3(64, Bn), 32, 0, stream>>>(VLAD, ASUM, nv_cw2, CNORM);
  zero_f32<<<1, 32, 0, stream>>>(GNORM, 32);
  vlad_intra<<<dim3(64, Bn), 32, 0, stream>>>(VLAD, CNORM, GNORM);
  vlad_global<<<dim3((unsigned)(((long long)32 * 65536 + 255) / 256)), 256, 0, stream>>>(
      VLAD, GNORM, VLADN);

  // ---- hidden GEMM: [32 x 65536] @ [65536 x 256], split-K atomic ----
  zero_f32<<<32, 256, 0, stream>>>(HID, 32 * 256);
  p = GemmParams{};
  p.A = VLADN; p.sAM = 65536; p.sAK = 1; p.sAB = 0;
  p.B = NVHWT; p.sBK = 1; p.sBN = 65536; p.sBB = 0;
  p.C = HID; p.sCM = 256; p.sCN = 1; p.sCB = 0;
  p.M = 32; p.N = 256; p.K = 65536; p.atomicAccum = 1;
  launch_gemm(stream, p, 1, 64);

  // ---- BN (eval), context gating, final L2 normalize ----
  bn_apply<<<32, 256, 0, stream>>>(HID, nv_bng, nv_bnb, VBN);
  gate_kernel<<<dim3(256, Bn), 32, 0, stream>>>(VBN, nv_gw, nv_gb, OPRE);
  final_norm<<<Bn, 32, 0, stream>>>(OPRE, out);
}
